// Compressor_86174223827251
// MI455X (gfx1250) — matrix operand order repaired and checked by reading, not compiled, not run on hardware
//
#include <hip/hip_runtime.h>
#include <hip/hip_bf16.h>

typedef __attribute__((ext_vector_type(16))) __bf16          v16bf;
typedef __attribute__((ext_vector_type(16))) unsigned short  v16u;
typedef __attribute__((ext_vector_type(8)))  float           v8f;

#define NDOWN_ELEMS (5*8*8*32*16)   // 163840 bf16 (w_down swizzled)
#define NSTOP_ELEMS (4*8*32*16)     // 16384  bf16 (w_stop swizzled)

__device__ __forceinline__ unsigned short f2bf(float f) {
    unsigned int u = __builtin_bit_cast(unsigned int, f);
    u += 0x7FFFu + ((u >> 16) & 1u);           // round-to-nearest-even
    return (unsigned short)(u >> 16);
}

// ---------------------------------------------------------------------------
// Prep: swizzle w_down / w_stop (f32, [out,in] layout) into the per-lane
// B-operand layout of v_wmma_f32_16x16x32_bf16:
//   B is KxN (N=16 cols per tile). lane = ((k/8)&1)*16 + n_local,
//   element v (0..15) within lane holds k32 = (lane/16)*8 + v (v<8), 16 + (lane/16)*8 + (v-8) (v>=8).
// Storage: [stage][kc][ntile][lane][16] bf16, w_stop appended after w_down.
// ---------------------------------------------------------------------------
__global__ __launch_bounds__(256) void compressor_prep(
    const float* __restrict__ wdown, const float* __restrict__ wstop,
    unsigned short* __restrict__ ws)
{
    int e = blockIdx.x * 256 + threadIdx.x;
    if (e >= NDOWN_ELEMS + NSTOP_ELEMS) return;
    int v  = e & 15;
    int L  = (e >> 4) & 31;
    int nt = (e >> 9) & 7;          // valid for both regions (NDOWN % 4096 == 0)
    int k32 = (v < 8) ? ((L >> 4) * 8 + v) : (16 + (L >> 4) * 8 + (v - 8));  // B-operand K within the 32-chunk
    int n   = nt * 16 + (L & 15);
    float val;
    if (e < NDOWN_ELEMS) {
        int kc = (e >> 12) & 7;
        int s  = e >> 15;
        val = wdown[((size_t)s * 128 + n) * 256 + (kc * 32 + k32)];
    } else {
        int e2 = e - NDOWN_ELEMS;
        int kc = e2 >> 12;          // 0..3
        val = wstop[(size_t)n * 128 + (kc * 32 + k32)];
    }
    ws[e] = f2bf(val);
}

// ---------------------------------------------------------------------------
// Main: one workgroup = 8 windows of one (b,h). 8 waves; wave owns tiles
// (m_tile,n_tile) round-robin. All 5 merge stages + stop projection via
// v_wmma_f32_16x16x32_bf16, ping-ponging bf16 activations through LDS.
// ---------------------------------------------------------------------------
__global__ __launch_bounds__(256) void compressor_main(
    const float* __restrict__ kin, const float* __restrict__ pe,
    const unsigned short* __restrict__ wdown_sw,
    const unsigned short* __restrict__ wstop_sw,
    float* __restrict__ out)
{
    extern __shared__ unsigned short smem[];
    unsigned short* Xa = smem;                 // 8*32*128 = 32768 bf16
    unsigned short* Xb = smem + 8 * 32 * 128;  // 8*16*128 = 16384 bf16

    const int tid  = threadIdx.x;
    const int wave = tid >> 5;
    const int lane = tid & 31;
    const int g    = lane >> 4;   // lane half (K-group select)
    const int lm   = lane & 15;   // row (A/D) or col (B/D) within tile

    const int wg      = blockIdx.x;
    const int bh      = wg >> 6;          // 0..31  (b*16+h)
    const int winBase = (wg & 63) * 8;    // first of 8 windows (0..504)

    // ---- stage-0 input: window gather of k (+pe), bf16 into LDS ----
    for (int i = 0; i < 128; ++i) {
        int e   = tid + i * 256;          // 8*32*128 = 32768 elements
        int win = e >> 12;
        int rem = e & 4095;
        int row = rem >> 7;
        int col = rem & 127;
        int w   = winBase + win;          // global window id; w==0 is zero block
        float v = 0.f;
        if (w > 0) {
            int srow = 16 * (w - 1) + row;
            v = kin[((size_t)bh * 8192 + srow) * 128 + col] + pe[row * 128 + col];
        }
        Xa[e] = f2bf(v);
    }
    __syncthreads();

    unsigned short* cur = Xa;
    unsigned short* nxt = Xb;

    // ---- 5 hierarchical merge stages: [aRows,256] x [256,128], SiLU ----
    for (int s = 0; s < 5; ++s) {
        const int inRows = 32 >> s;           // rows/window in cur
        const int aRows  = 16 >> s;           // A rows (= out rows) per window
        const int shift  = 4 - s;             // log2(aRows)
        const int totalA = 8 << shift;        // valid A rows across 8 windows
        const int mTiles = (totalA + 15) >> 4;
        const unsigned short* wbase = wdown_sw + s * (8 * 8 * 32 * 16);

        for (int t = wave; t < mTiles * 8; t += 8) {
            const int mt = t >> 3;
            const int nt = t & 7;
            v8f acc = {0.f, 0.f, 0.f, 0.f, 0.f, 0.f, 0.f, 0.f};

            // A row handled by this lane (documented 16-bit A layout: M = lane%16)
            int gm  = mt * 16 + lm;
            int win = gm >> shift;
            int am  = gm & ((1 << shift) - 1);
            if (win > 7) { win = 0; am = 0; }       // clamp (store is guarded)
            const unsigned short* abase = cur + ((size_t)(win * inRows + am * 2)) * 128;

            for (int kc = 0; kc < 8; ++kc) {
                v16u av;
                #pragma unroll
                for (int vg = 0; vg < 8; ++vg) {
                    int k32 = (vg < 4) ? (g * 8 + 2 * vg)
                                       : (16 + g * 8 + 2 * (vg - 4));
                    unsigned int u =
                        *reinterpret_cast<const unsigned int*>(abase + kc * 32 + k32);
                    av[2 * vg]     = (unsigned short)(u & 0xFFFFu);
                    av[2 * vg + 1] = (unsigned short)(u >> 16);
                }
                const unsigned short* bptr = wbase + (((kc * 8 + nt) * 32) + lane) * 16;
                v16u bv = *reinterpret_cast<const v16u*>(bptr);

                acc = __builtin_amdgcn_wmma_f32_16x16x32_bf16(
                        false, __builtin_bit_cast(v16bf, av),
                        false, __builtin_bit_cast(v16bf, bv),
                        (short)0, acc, false, false);
            }

            // SiLU + scatter D (row = r + 8*g, col = lm) to next LDS buffer
            #pragma unroll
            for (int r = 0; r < 8; ++r) {
                int m   = r + 8 * g;
                int ogm = mt * 16 + m;
                if (ogm < totalA) {
                    int owin = ogm >> shift;
                    int orow = ogm & ((1 << shift) - 1);
                    float d  = acc[r];
                    float sv = d / (1.f + __expf(-d));
                    nxt[((size_t)(owin << shift) + orow) * 128 + nt * 16 + lm] = f2bf(sv);
                }
            }
        }
        __syncthreads();
        unsigned short* tmp = cur; cur = nxt; nxt = tmp;
    }

    // ---- stop projection: [8,128] x [128,128] -> d_out (f32, no SiLU) ----
    {
        const int nt = wave;                  // 8 waves, 8 N-tiles, 1 M-tile
        v8f acc = {0.f, 0.f, 0.f, 0.f, 0.f, 0.f, 0.f, 0.f};
        int win = (lm < 8) ? lm : 0;          // A row = window index (8 valid)
        const unsigned short* abase = cur + (size_t)win * 128;

        for (int kc = 0; kc < 4; ++kc) {
            v16u av;
            #pragma unroll
            for (int vg = 0; vg < 8; ++vg) {
                int k32 = (vg < 4) ? (g * 8 + 2 * vg)
                                   : (16 + g * 8 + 2 * (vg - 4));
                unsigned int u =
                    *reinterpret_cast<const unsigned int*>(abase + kc * 32 + k32);
                av[2 * vg]     = (unsigned short)(u & 0xFFFFu);
                av[2 * vg + 1] = (unsigned short)(u >> 16);
            }
            const unsigned short* bptr = wstop_sw + (((kc * 8 + nt) * 32) + lane) * 16;
            v16u bv = *reinterpret_cast<const v16u*>(bptr);

            acc = __builtin_amdgcn_wmma_f32_16x16x32_bf16(
                    false, __builtin_bit_cast(v16bf, av),
                    false, __builtin_bit_cast(v16bf, bv),
                    (short)0, acc, false, false);
        }

        #pragma unroll
        for (int r = 0; r < 8; ++r) {
            int m = r + 8 * g;               // output row = window index
            if (m < 8) {
                out[(((size_t)bh * 512) + winBase + m) * 128 + nt * 16 + lm] = acc[r];
            }
        }
    }
}

extern "C" void kernel_launch(void* const* d_in, const int* in_sizes, int n_in,
                              void* d_out, int out_size, void* d_ws, size_t ws_size,
                              hipStream_t stream) {
    const float* k     = (const float*)d_in[0];   // [2,16,8192,128]
    const float* pe    = (const float*)d_in[1];   // [32,128]
    const float* wdown = (const float*)d_in[2];   // [5,128,256]
    const float* wstop = (const float*)d_in[3];   // [128,128]
    float* out = (float*)d_out;                   // [2,16,512,128]
    unsigned short* ws = (unsigned short*)d_ws;

    const int prepN = NDOWN_ELEMS + NSTOP_ELEMS;
    compressor_prep<<<(prepN + 255) / 256, 256, 0, stream>>>(wdown, wstop, ws);

    const size_t shmem = (size_t)(8 * 32 * 128 + 8 * 16 * 128) * sizeof(unsigned short); // 96 KB
    compressor_main<<<dim3(2048), dim3(256), shmem, stream>>>(
        k, pe, ws, ws + NDOWN_ELEMS, out);
}
